// crossMultiHead_14542759264516
// MI455X (gfx1250) — matrix operand order repaired and checked by reading, not compiled, not run on hardware
//
#include <hip/hip_runtime.h>

// ---------------------------------------------------------------------------
// MI455X / gfx1250 implementation.
//
//   Q = a @ W_Q^T, K = a @ W_K^T  : M=2048, N=4096, K=4096 bf16 WMMA GEMMs
//   scores = Q^T K / sqrt(512), softmax, context = leakyrelu(V @ attn)
//
// Workspace layout (all bf16 stored as ushort):
//   a_bf   : 2048*4096   | wq_bf/wk_bf : 4096*4096 each
//   Qt/Kt  : 256*64*512  ([n][row][d], d contiguous)
//   Vbf    : 32*4096*64  ([b][s][t], t contiguous)      total 128 MiB
// ---------------------------------------------------------------------------

typedef __attribute__((ext_vector_type(16))) __bf16 v16bf;
typedef __attribute__((ext_vector_type(8)))  float  v8f;

union Frag {
  uint4 q[2];   // 2 x 16B = 32B = 16 bf16
  v16bf v;
};

__device__ __forceinline__ unsigned short f2bf(float f) {
  union { float f; unsigned int u; } c; c.f = f;
  unsigned int r = c.u + 0x7FFFu + ((c.u >> 16) & 1u);  // round-to-nearest-even
  return (unsigned short)(r >> 16);
}

__device__ __forceinline__ v8f wmma_bf16(const Frag& a, const Frag& b, v8f c) {
  return __builtin_amdgcn_wmma_f32_16x16x32_bf16(
      /*neg_a=*/false, a.v, /*neg_b=*/false, b.v,
      /*c_mod=*/(short)0, c, /*reuse_a=*/false, /*reuse_b=*/false);
}

// ---------------------------------------------------------------- convert ---
__global__ void cvt_f32_bf16(const float* __restrict__ in,
                             unsigned short* __restrict__ out, int n4) {
  int i = blockIdx.x * blockDim.x + threadIdx.x;
  if (i < n4) {
    float4 f = ((const float4*)in)[i];
    union { unsigned short s[4]; uint2 u; } o;
    o.s[0] = f2bf(f.x); o.s[1] = f2bf(f.y);
    o.s[2] = f2bf(f.z); o.s[3] = f2bf(f.w);
    ((uint2*)out)[i] = o.u;
  }
}

// --------------------------------------------------------- V projection -----
// Vbf[b][s][t] = bf16( sum_c x[b][c][s] * Wv[t][c] )   (small GEMM, VALU)
__global__ __launch_bounds__(256) void vproj(const float* __restrict__ x,
                                             const float* __restrict__ Wv,
                                             unsigned short* __restrict__ Vbf) {
  __shared__ float xs[64][65];   // xs[c][i]
  __shared__ float wv[64][65];   // wv[t][c]
  const int b  = blockIdx.y;
  const int s0 = blockIdx.x * 64;
  const int tid = threadIdx.x;
  {
    int c = tid >> 2, i0 = (tid & 3) * 16;
    const float* gp = x + (((size_t)b * 64 + c) << 12) + s0 + i0;
    for (int k = 0; k < 16; ++k) xs[c][i0 + k] = gp[k];
    const float* wp = Wv + c * 64 + i0;   // reuse (c,i0) as (t,c0)
    for (int k = 0; k < 16; ++k) wv[c][i0 + k] = wp[k];
  }
  __syncthreads();
  const int i  = tid & 63;
  const int t0 = (tid >> 6) * 16;
  for (int t = t0; t < t0 + 16; ++t) {
    float s = 0.f;
    for (int c = 0; c < 64; ++c) s = fmaf(xs[c][i], wv[t][c], s);
    Vbf[(((size_t)b << 12) + s0 + i) * 64 + t] = f2bf(s);
  }
}

// ------------------------------------------------------- Q/K projections ----
// C[m][t] = sum_s a_bf[m][s] * W[t][s];  epilogue scatters into Qt/Kt layout.
// Workgroup tile 128(M) x 256(N), K-step 32, LDS double-buffered.
// 8 waves in 2x4 grid; each wave computes 64x64 (16 WMMA tiles).
#define LDK 40   // padded LDS k-stride (ushorts); 80B rows, 16B aligned
#define A_BUF (128 * LDK)
#define B_BUF (256 * LDK)
__global__ __launch_bounds__(256) void qk_gemm(
    const unsigned short* __restrict__ Abf,
    const unsigned short* __restrict__ Wq,
    const unsigned short* __restrict__ Wk,
    unsigned short* __restrict__ Qt,
    unsigned short* __restrict__ Kt) {
  const unsigned short* __restrict__ W = (blockIdx.z == 0) ? Wq : Wk;
  unsigned short* __restrict__ Out     = (blockIdx.z == 0) ? Qt : Kt;

  __shared__ __attribute__((aligned(16))) unsigned short As[2 * A_BUF];
  __shared__ __attribute__((aligned(16))) unsigned short Bs[2 * B_BUF];

  const int tid  = threadIdx.x;
  const int lane = tid & 31;
  const int wv   = tid >> 5;
  const int wm   = wv & 1;          // M block of 64
  const int wn   = wv >> 1;         // N block of 64
  const int m0   = blockIdx.y * 128;
  const int n0   = blockIdx.x * 256;

  // staging pattern: thread covers A row (tid>>1) and B rows (tid>>1), (tid>>1)+128,
  // 16-element half h = (tid&1)*16 of the 32-wide K slice.
  const int srow = tid >> 1;
  const int sh   = (tid & 1) * 16;
  const unsigned short* gA  = Abf + (size_t)(m0 + srow) * 4096 + sh;
  const unsigned short* gB0 = W   + (size_t)(n0 + srow) * 4096 + sh;
  const unsigned short* gB1 = gB0 + (size_t)128 * 4096;
  unsigned short* sA  = &As[srow * LDK + sh];
  unsigned short* sB0 = &Bs[srow * LDK + sh];
  unsigned short* sB1 = sB0 + 128 * LDK;

  v8f acc[4][4];
  for (int i = 0; i < 4; ++i)
    for (int j = 0; j < 4; ++j)
      for (int r = 0; r < 8; ++r) acc[i][j][r] = 0.f;

  // ---- prologue: stage k-slice 0 into buffer 0
  {
    uint4 a0 = ((const uint4*)gA)[0],  a1 = ((const uint4*)gA)[1];
    uint4 b0 = ((const uint4*)gB0)[0], b1 = ((const uint4*)gB0)[1];
    uint4 b2 = ((const uint4*)gB1)[0], b3 = ((const uint4*)gB1)[1];
    *(uint4*)sA = a0;        *(uint4*)(sA + 8) = a1;
    *(uint4*)sB0 = b0;       *(uint4*)(sB0 + 8) = b1;
    *(uint4*)sB1 = b2;       *(uint4*)(sB1 + 8) = b3;
  }
  __syncthreads();

  int buf = 0;
  for (int kt = 0; kt < 4096; kt += 32) {
    const bool has_next = (kt + 32) < 4096;
    uint4 a0, a1, b0, b1, b2, b3;
    if (has_next) {
      const int kn = kt + 32;
      a0 = ((const uint4*)(gA + kn))[0];  a1 = ((const uint4*)(gA + kn))[1];
      b0 = ((const uint4*)(gB0 + kn))[0]; b1 = ((const uint4*)(gB0 + kn))[1];
      b2 = ((const uint4*)(gB1 + kn))[0]; b3 = ((const uint4*)(gB1 + kn))[1];
      if (kt + 96 < 4096) {   // pull the streaming tiles early (global_prefetch_b8)
        __builtin_prefetch((const void*)(gA + kt + 96), 0, 0);
        __builtin_prefetch((const void*)(gB0 + kt + 96), 0, 0);
        __builtin_prefetch((const void*)(gB1 + kt + 96), 0, 0);
      }
    }

    // ---- compute current buffer: hoist all 8 fragments, then 16 WMMA burst
    {
      const unsigned short* Ab = &As[buf * A_BUF];
      const unsigned short* Bb = &Bs[buf * B_BUF];
      Frag afr[4], bfr[4];
      for (int i = 0; i < 4; ++i) {
        const unsigned short* p =
            &Ab[(wm * 64 + i * 16 + (lane & 15)) * LDK + ((lane >> 4) << 4)];
        afr[i].q[0] = *(const uint4*)p;          // per-lane K chunk 16h..16h+7
        afr[i].q[1] = *(const uint4*)(p + 8);    // per-lane K chunk 16h+8..16h+15
      }
      for (int j = 0; j < 4; ++j) {
        const unsigned short* p =
            &Bb[(wn * 64 + j * 16 + (lane & 15)) * LDK + ((lane >> 4) << 4)];
        bfr[j].q[0] = *(const uint4*)p;
        bfr[j].q[1] = *(const uint4*)(p + 8);
      }
      for (int i = 0; i < 4; ++i)
        for (int j = 0; j < 4; ++j)
          acc[i][j] = wmma_bf16(afr[i], bfr[j], acc[i][j]);
    }

    if (has_next) {
      unsigned short* dA  = sA  + (buf ^ 1) * A_BUF;
      unsigned short* dB0 = sB0 + (buf ^ 1) * B_BUF;
      unsigned short* dB1 = sB1 + (buf ^ 1) * B_BUF;
      *(uint4*)dA = a0;        *(uint4*)(dA + 8) = a1;
      *(uint4*)dB0 = b0;       *(uint4*)(dB0 + 8) = b1;
      *(uint4*)dB1 = b2;       *(uint4*)(dB1 + 8) = b3;
    }
    __syncthreads();
    buf ^= 1;
  }

  // Epilogue: Q_raw[b][c][t] -> Qt[n][e][d], n=b*8+c/8, d=(c%8)*64+t/64, e=t%64
  for (int i = 0; i < 4; ++i)
    for (int j = 0; j < 4; ++j)
      for (int r = 0; r < 8; ++r) {
        int m = m0 + wm * 64 + i * 16 + r + ((lane >> 4) << 3);
        int t = n0 + wn * 64 + j * 16 + (lane & 15);
        int b = m >> 6, c = m & 63;
        int nh = (b << 3) + (c >> 3);
        int d  = ((c & 7) << 6) + (t >> 6);
        int e  = t & 63;
        Out[(((size_t)(nh * 64 + e)) << 9) + d] = f2bf(acc[i][j][r]);
      }
}

// ---------------------------------------- scores + softmax + context --------
#define LDQ 72   // attn^T LDS k-stride (ushorts); 144B rows, 16B aligned
__global__ __launch_bounds__(256) void attn_ctx(
    const unsigned short* __restrict__ Qt,
    const unsigned short* __restrict__ Kt,
    const unsigned short* __restrict__ Vbf,
    float* __restrict__ Out) {
  __shared__ float sc[64][66];
  __shared__ __attribute__((aligned(16))) unsigned short at[64 * LDQ];
  const int n    = blockIdx.x;
  const int tid  = threadIdx.x;
  const int lane = tid & 31;
  const int wv   = tid >> 5;

  // ---- scores[c][e] = (1/sqrt(512)) * sum_d Qt[n][c][d] * Kt[n][e][d]
  {
    const int ct = wv >> 1;          // c tile 0..3
    const int e0 = (wv & 1) * 2;     // e tiles e0, e0+1
    v8f acc[2];
    for (int j = 0; j < 2; ++j)
      for (int r = 0; r < 8; ++r) acc[j][r] = 0.f;
    const unsigned short* qb =
        Qt + ((size_t)(n * 64 + ct * 16 + (lane & 15)) << 9);
    const unsigned short* kb0 =
        Kt + ((size_t)(n * 64 + e0 * 16 + (lane & 15)) << 9);
    const unsigned short* kb1 = kb0 + ((size_t)16 << 9);
    for (int ks = 0; ks < 512; ks += 32) {
      Frag af, b0, b1;
      int ka = ks + ((lane >> 4) << 4);
      af.q[0] = *(const uint4*)(qb + ka);
      af.q[1] = *(const uint4*)(qb + ka + 8);
      int kb = ks + ((lane >> 4) << 4);
      b0.q[0] = *(const uint4*)(kb0 + kb);
      b0.q[1] = *(const uint4*)(kb0 + kb + 8);
      b1.q[0] = *(const uint4*)(kb1 + kb);
      b1.q[1] = *(const uint4*)(kb1 + kb + 8);
      acc[0] = wmma_bf16(af, b0, acc[0]);
      acc[1] = wmma_bf16(af, b1, acc[1]);
    }
    const float scale = 0.044194173824159216f;  // 1/sqrt(512)
    for (int j = 0; j < 2; ++j)
      for (int r = 0; r < 8; ++r) {
        int c = ct * 16 + r + ((lane >> 4) << 3);
        int e = (e0 + j) * 16 + (lane & 15);
        sc[c][e] = acc[j][r] * scale;
      }
  }
  __syncthreads();

  // ---- softmax over e; write attn^T (bf16) to LDS: at[e][q]
  if (tid < 64) {
    float mx = -1e30f;
    for (int e = 0; e < 64; ++e) mx = fmaxf(mx, sc[tid][e]);
    float sum = 0.f;
    for (int e = 0; e < 64; ++e) sum += __expf(sc[tid][e] - mx);
    float inv = 1.f / sum;
    for (int e = 0; e < 64; ++e)
      at[e * LDQ + tid] = f2bf(__expf(sc[tid][e] - mx) * inv);
  }
  __syncthreads();

  // ---- context[p][e] = sum_q Vbf[n*512+p][q] * attn[q][e]; leaky relu
  {
    Frag bf[2][4];
    for (int kq = 0; kq < 2; ++kq)
      for (int et = 0; et < 4; ++et) {
        const unsigned short* p =
            &at[(et * 16 + (lane & 15)) * LDQ + kq * 32 + ((lane >> 4) << 4)];
        bf[kq][et].q[0] = *(const uint4*)p;
        bf[kq][et].q[1] = *(const uint4*)(p + 8);
      }
    const unsigned short* vb = Vbf + (((size_t)n) << 9) * 64;  // (n*512)*64
    float* ob = Out + ((size_t)n << 15);
    for (int i = 0; i < 4; ++i) {
      const int p0 = wv * 64 + i * 16;
      v8f acc[4];
      for (int et = 0; et < 4; ++et)
        for (int r = 0; r < 8; ++r) acc[et][r] = 0.f;
      for (int kq = 0; kq < 2; ++kq) {
        Frag af;
        const unsigned short* ap =
            vb + (size_t)(p0 + (lane & 15)) * 64 + kq * 32 + ((lane >> 4) << 4);
        af.q[0] = *(const uint4*)ap;
        af.q[1] = *(const uint4*)(ap + 8);
        for (int et = 0; et < 4; ++et)
          acc[et] = wmma_bf16(af, bf[kq][et], acc[et]);
      }
      for (int et = 0; et < 4; ++et)
        for (int r = 0; r < 8; ++r) {
          float v = acc[et][r];
          v = (v > 0.f) ? v : 0.2f * v;
          int p = p0 + r + ((lane >> 4) << 3);
          int e = et * 16 + (lane & 15);
          ob[(size_t)e * 512 + p] = v;
        }
    }
  }
}

// ---------------------------------------------------------------------------
extern "C" void kernel_launch(void* const* d_in, const int* in_sizes, int n_in,
                              void* d_out, int out_size, void* d_ws, size_t ws_size,
                              hipStream_t stream) {
  const float* z  = (const float*)d_in[0];   // 32*64*64*64
  const float* x  = (const float*)d_in[1];   // 32*64*64*64
  const float* Wq = (const float*)d_in[2];   // 4096*4096
  const float* Wk = (const float*)d_in[3];   // 4096*4096
  const float* Wv = (const float*)d_in[4];   // 64*64
  float* out = (float*)d_out;                // 32*64*64*64

  unsigned short* a_bf  = (unsigned short*)d_ws;
  unsigned short* wq_bf = a_bf  + (size_t)2048 * 4096;
  unsigned short* wk_bf = wq_bf + (size_t)4096 * 4096;
  unsigned short* qt    = wk_bf + (size_t)4096 * 4096;
  unsigned short* kt    = qt    + (size_t)256 * 64 * 512;
  unsigned short* vbf   = kt    + (size_t)256 * 64 * 512;

  const int nA = 2048 * 4096 / 4;
  const int nW = 4096 * 4096 / 4;
  cvt_f32_bf16<<<(nA + 255) / 256, 256, 0, stream>>>(z,  a_bf,  nA);
  cvt_f32_bf16<<<(nW + 255) / 256, 256, 0, stream>>>(Wq, wq_bf, nW);
  cvt_f32_bf16<<<(nW + 255) / 256, 256, 0, stream>>>(Wk, wk_bf, nW);
  vproj<<<dim3(64, 32), 256, 0, stream>>>(x, Wv, vbf);
  qk_gemm<<<dim3(16, 16, 2), 256, 0, stream>>>(a_bf, wq_bf, wk_bf, qt, kt);
  attn_ctx<<<256, 256, 0, stream>>>(qt, kt, vbf, out);
}
